// Evaluation_HIoU_90452011253876
// MI455X (gfx1250) — 8-bit matrix operand order repaired and checked by reading, compiled, not run on hardware
//
#include <hip/hip_runtime.h>
#include <stdint.h>

#define HH 400
#define WW 400
#define NPIX (HH * WW)
#define NPIECE 256
#define CHUNK 64
#define NCHUNK (NPIX / CHUNK)
#define P2_WAVES 8

typedef int v8i __attribute__((ext_vector_type(8)));


__global__ void zero_counts(unsigned* __restrict__ aP, unsigned* __restrict__ aG) {
  int t = blockIdx.x * blockDim.x + threadIdx.x;
  if (t < NPIECE) { aP[t] = 0u; aG[t] = 0u; }
}

__device__ __forceinline__ int piece_index(const float* __restrict__ pts, float fx, float fy) {
  int idx = 0;
  bool valid = true;
#pragma unroll
  for (int i = 0; i < 8; ++i) {
    float x1 = pts[4*i+0], y1 = pts[4*i+1], x2 = pts[4*i+2], y2 = pts[4*i+3];
    bool vert = (x1 == x2);
    float a = (y1 - y2) / (vert ? 1.0f : (x1 - x2));
    float c = -a * x1 + y1;
    float d = vert ? (fx - x1) : (fx * a - fy + c);
    if (d < 0.0f)            idx |= (1 << i);
    else if (!(d > 0.0f))    valid = false;
  }
  return valid ? idx : -1;
}

__global__ void __launch_bounds__(256)
phase1_classify(const float* __restrict__ pred, const float* __restrict__ gt,
                short* __restrict__ mp, short* __restrict__ mg,
                unsigned* __restrict__ aP, unsigned* __restrict__ aG) {
  int p = blockIdx.x * blockDim.x + threadIdx.x;
  if (p >= NPIX) return;
  float fx = (float)(p % WW);
  float fy = (float)(p / WW);
  int ip = piece_index(pred, fx, fy);
  int ig = piece_index(gt,   fx, fy);
  mp[p] = (short)ip;
  mg[p] = (short)ig;
  if (ip >= 0) atomicAdd(&aP[ip], 1u);
  if (ig >= 0) atomicAdd(&aG[ig], 1u);
}

__device__ __forceinline__ unsigned pack4w(unsigned lo, unsigned hi, int t) {
  int s0 = (short)(lo & 0xffffu);
  int s1 = (short)(lo >> 16);
  int s2 = (short)(hi & 0xffffu);
  int s3 = (short)(hi >> 16);
  unsigned r = (s0 == t) ? 0x00000001u : 0u;
  if (s1 == t) r |= 0x00000100u;
  if (s2 == t) r |= 0x00010000u;
  if (s3 == t) r |= 0x01000000u;
  return r;
}

__device__ __forceinline__ void async_stage_b32(unsigned lds_off, const void* gptr) {
  asm volatile("global_load_async_to_lds_b32 %0, %1, off"
               :: "v"(lds_off), "v"(gptr)
               : "memory");
}
__device__ __forceinline__ void wait_asynccnt0() {
  asm volatile("s_wait_asynccnt 0x0" ::: "memory");
}

__global__ void __launch_bounds__(256)
phase2_wmma(const short* __restrict__ mp, const short* __restrict__ mg,
            int* __restrict__ inter) {
  __shared__ __align__(16) short mpS[2][CHUNK];
  __shared__ __align__(16) short mgS[2][CHUNK];

  const int t     = threadIdx.x;
  const int lane  = t & 31;
  const int wave  = t >> 5;
  const int tile  = blockIdx.x * P2_WAVES + wave;
  const int m0    = (tile >> 4) * 16;
  const int n0    = (tile & 15) * 16;
  const int lhalf = lane >> 4;
  const int lrow  = lane & 15;
  const int atgt  = m0 + lrow;
  const int btgt  = n0 + lrow;
  const int ahi   = lhalf * 8;
  const int bhi   = lhalf * 16;

  const unsigned mpOff0 = (unsigned)(uintptr_t)&mpS[0][0] + 4u * (unsigned)lane;
  const unsigned mpOff1 = (unsigned)(uintptr_t)&mpS[1][0] + 4u * (unsigned)lane;
  const unsigned mgOff0 = (unsigned)(uintptr_t)&mgS[0][0] + 4u * (unsigned)lane;
  const unsigned mgOff1 = (unsigned)(uintptr_t)&mgS[1][0] + 4u * (unsigned)lane;

  const int* mpw = (const int*)mp;
  const int* mgw = (const int*)mg;

  v8i acc = {0, 0, 0, 0, 0, 0, 0, 0};

  if (wave == 0)      async_stage_b32(mpOff0, mpw + lane);
  else if (wave == 1) async_stage_b32(mgOff0, mgw + lane);

  for (int c = 0; c < NCHUNK; ++c) {
    wait_asynccnt0();
    __syncthreads();
    const int cur = c & 1;

    if (c + 1 < NCHUNK) {
      if (wave == 0) {
        async_stage_b32(cur ? mpOff0 : mpOff1, mpw + (c + 1) * 32 + lane);
        __builtin_prefetch(mpw + (c + 8) * 32 + lane, 0, 1);
      } else if (wave == 1) {
        async_stage_b32(cur ? mgOff0 : mgOff1, mgw + (c + 1) * 32 + lane);
        __builtin_prefetch(mgw + (c + 8) * 32 + lane, 0, 1);
      }
    }

    v8i A, B;
#pragma unroll
    for (int j = 0; j < 8; ++j) {
      int kb = ((j >> 1) * 16) + ((j & 1) * 4) + ahi;
      unsigned lo = *(const unsigned*)&mpS[cur][kb];
      unsigned hi = *(const unsigned*)&mpS[cur][kb + 2];
      A[j] = (int)pack4w(lo, hi, atgt);
    }
#pragma unroll
    for (int j = 0; j < 8; ++j) {
      int kb = ((j >> 1) * 16) + ((j & 1) * 4) + ahi;
      unsigned lo = *(const unsigned*)&mgS[cur][kb];
      unsigned hi = *(const unsigned*)&mgS[cur][kb + 2];
      B[j] = (int)pack4w(lo, hi, btgt);
    }

    acc = __builtin_amdgcn_wmma_i32_16x16x64_iu8(false, A, false, B, acc, false, false);
  }

  const int col   = n0 + lrow;
  const int rbase = m0 + lhalf * 8;
#pragma unroll
  for (int r = 0; r < 8; ++r)
    inter[(rbase + r) * NPIECE + col] = acc[r];
}

__global__ void __launch_bounds__(256)
phase3_hiou(const unsigned* __restrict__ aP, const unsigned* __restrict__ aG,
            const int* __restrict__ inter, float* __restrict__ out) {
  __shared__ float aPs[NPIECE], aGs[NPIECE];
  __shared__ float redN[NPIECE], redD[NPIECE];
  const int t = threadIdx.x;
  const float ap = (float)aP[t];
  const float ag = (float)aG[t];
  aPs[t] = ap; aGs[t] = ag;
  __syncthreads();

  float rmax = 0.0f;
  for (int n = 0; n < NPIECE; ++n) {
    float it  = (float)inter[t * NPIECE + n];
    float uni = ap + aGs[n] - it;
    float iou = (uni > 0.0f) ? (it / uni) : 0.0f;
    rmax = fmaxf(rmax, iou);
  }
  float cmax = 0.0f;
  for (int m = 0; m < NPIECE; ++m) {
    float it  = (float)inter[m * NPIECE + t];
    float uni = aPs[m] + ag - it;
    float iou = (uni > 0.0f) ? (it / uni) : 0.0f;
    cmax = fmaxf(cmax, iou);
  }

  float vp = (ap > 0.0f) ? 1.0f : 0.0f;
  float vg = (ag > 0.0f) ? 1.0f : 0.0f;
  redN[t] = rmax * vp + cmax * vg;
  redD[t] = vp + vg;
  __syncthreads();
  for (int s = NPIECE / 2; s > 0; s >>= 1) {
    if (t < s) { redN[t] += redN[t + s]; redD[t] += redD[t + s]; }
    __syncthreads();
  }
  if (t == 0) out[0] = (redD[0] > 0.0f) ? (redN[0] / redD[0]) : 0.0f;
}

extern "C" void kernel_launch(void* const* d_in, const int* in_sizes, int n_in,
                              void* d_out, int out_size, void* d_ws, size_t ws_size,
                              hipStream_t stream) {
  const float* pred = (const float*)d_in[0];
  const float* gt   = (const float*)d_in[1];
  char* ws = (char*)d_ws;
  short*    mp    = (short*)   (ws + 0);
  short*    mg    = (short*)   (ws + 320000);
  unsigned* aP    = (unsigned*)(ws + 640000);
  unsigned* aG    = (unsigned*)(ws + 641024);
  int*      inter = (int*)     (ws + 642048);
  float*    out   = (float*)d_out;

  zero_counts<<<1, 256, 0, stream>>>(aP, aG);
  phase1_classify<<<(NPIX + 255) / 256, 256, 0, stream>>>(pred, gt, mp, mg, aP, aG);
  phase2_wmma<<<NPIECE / P2_WAVES, 256, 0, stream>>>(mp, mg, inter);
  phase3_hiou<<<1, 256, 0, stream>>>(aP, aG, inter, out);
}
